// ESNModule_12403865551690
// MI455X (gfx1250) — matrix operand order repaired and checked by reading, not compiled, not run on hardware
//
#include <hip/hip_runtime.h>
#include <hip/hip_bf16.h>
#include <math.h>

// ---------------------------------------------------------------------------
// ESN forward: states[b, t+1, :] = (1-a)*s + a*tanh(s @ w_res^T + X[:,t,:] @ w_in^T)
// B=256, T=1024, IN=256, OUT=1024, alpha=0.1
// Per-step fused GEMM over K = OUT+IN = 1280 with v_wmma_f32_16x16x32_bf16,
// fp32 accumulate. B (stacked [w_res^T; w_in^T]) pre-swizzled to fragment
// order and staged block-wide into LDS with global_load_async_to_lds_b128,
// double-buffered (async copy of chunk c+1 overlaps WMMAs of chunk c).
// State double-buffered in WMMA A-fragment order across step kernels.
// Workspace requirement: ~4.5 MB.
// ---------------------------------------------------------------------------

typedef __bf16 v16bf __attribute__((ext_vector_type(16)));
typedef float  v8f   __attribute__((ext_vector_type(8)));

#define B_DIM   256
#define T_DIM   1024
#define IN_DIM  256
#define OUT_DIM 1024
#define NT_M    16      // B_DIM/16   (C row tiles)
#define NT_N    64      // OUT_DIM/16 (C col tiles)
#define NKT_S   32      // OUT_DIM/32 (state K tiles)
#define NKT_X   8       // IN_DIM/32  (input K tiles)
#define NKT     40      // total K tiles (K = 1280)
#define CHUNK   8       // K tiles per LDS stage (5 chunks: 4 state + 1 input)
#define ALPHA   0.1f

// Workspace layout (bytes). Total = 4,718,592 B (~4.5 MB).
#define BP_OFF    0u                       // [NT_N][NKT][32][16] ushort = 2,621,440 B
#define SBF0_OFF  2621440u                 // [NT_M][NKT_S][32][16] ushort = 524,288 B
#define SBF1_OFF  3145728u                 // ditto
#define SF32_OFF  3670016u                 // [B_DIM][OUT_DIM] float = 1,048,576 B

__device__ __forceinline__ unsigned short f32_to_bf16_rne(float f) {
    union { float f; unsigned int u; } c; c.f = f;
    unsigned int u = c.u;
    unsigned int r = u + 0x7FFFu + ((u >> 16) & 1u);   // round-to-nearest-even
    return (unsigned short)(r >> 16);
}

// Async global->LDS copy, 16 B per lane (CDNA5, ASYNCcnt-tracked).
// lds_off: LDS byte offset (generic shared addr truncated to 32 bits, per
// ISA aperture rule).
__device__ __forceinline__ void async_b128(unsigned int lds_off, const void* gaddr) {
    asm volatile("global_load_async_to_lds_b128 %0, %1, off"
                 :: "v"(lds_off), "v"(gaddr)
                 : "memory");
}

__device__ __forceinline__ void wait_async0() {
    asm volatile("s_wait_asynccnt 0x0" ::: "memory");
}

// --- init: zero fp32 state, zero bf16 state buffer 0, zero out[:,0,:] -------
__global__ __launch_bounds__(256) void esn_init(float* __restrict__ Sf32,
                                                unsigned short* __restrict__ Sbf0,
                                                float* __restrict__ out) {
    int idx = blockIdx.x * 256 + threadIdx.x;          // 0 .. 262143
    int m = idx >> 10;
    int o = idx & 1023;
    Sf32[idx] = 0.0f;
    Sbf0[idx] = 0;                                     // bf16 zero (layout-agnostic)
    out[(size_t)m * (size_t)((T_DIM + 1) * OUT_DIM) + o] = 0.0f;  // states[:,0,:]=0
}

// --- pack stacked B = [w_res^T ; w_in^T] into WMMA B-fragment order ---------
// B[k][n] = (k < 1024) ? w_res[n][k] : w_in[n][k-1024]
// Fragment: lane = n_local + 16*half holds k_local = 8*half + j (j < 8), 16 + 8*half + (j-8) (j >= 8).
__global__ __launch_bounds__(256) void esn_pack_b(const float* __restrict__ w_res,
                                                  const float* __restrict__ w_in,
                                                  unsigned short* __restrict__ Bp) {
    int tn  = blockIdx.y;                              // 0..63
    int tid = blockIdx.x * 256 + threadIdx.x;          // 0..20479
    int j    = tid & 15;
    int lane = (tid >> 4) & 31;
    int tk   = tid >> 9;                               // 0..39
    int n    = tn * 16 + (lane & 15);
    int half = lane >> 4;
    int k    = tk * 32 + half * 8 + (j & 7) + ((j >> 3) << 4);
    float v = (k < OUT_DIM) ? w_res[(size_t)n * OUT_DIM + k]
                            : w_in [(size_t)n * IN_DIM + (k - OUT_DIM)];
    Bp[(((size_t)tn * NKT + tk) * 32 + lane) * 16 + j] = f32_to_bf16_rne(v);
}

// --- one recurrence step ----------------------------------------------------
// One wave per 16x16 C tile; 4 waves/block share tile_n. B fragments for the
// whole block are staged into LDS by async DMA, double buffered by chunk.
__global__ __launch_bounds__(128) void esn_step(const float* __restrict__ X,
                                                const unsigned short* __restrict__ Bp,
                                                const unsigned short* __restrict__ SbfRd,
                                                unsigned short* __restrict__ SbfWr,
                                                float* __restrict__ Sf32,
                                                float* __restrict__ out,
                                                int t) {
    __shared__ __align__(64) unsigned short shB[2][CHUNK * 512];  // 2 x 8 KB

    const int tid  = threadIdx.x;
    const int lane = tid & 31;
    const int wid  = blockIdx.x * 4 + (tid >> 5);      // tile id 0..1023
    const int tm   = wid & 15;                         // C row tile
    const int tn   = wid >> 4;                         // C col tile (same for block)
    const int half = lane >> 4;
    const int nn   = lane & 15;

    const unsigned short* BpBlk = Bp + (size_t)tn * NKT * 512;     // this block's B
    const unsigned short* Ar = SbfRd + ((size_t)tm * NKT_S * 32 + lane) * 16;
    const float* px = X + ((size_t)(tm * 16 + nn) * T_DIM + t) * IN_DIM + half * 8;

    const unsigned int shb_base0 = (unsigned int)(size_t)&shB[0][0];
    const unsigned int shb_base1 = (unsigned int)(size_t)&shB[1][0];

    // ---- stage chunk 0 (128 threads x 4 x 16 B = 8 KB) ----
    {
        const char* s = (const char*)BpBlk;
        #pragma unroll
        for (int q = 0; q < 4; ++q) {
            int e = q * 128 + tid;
            async_b128(shb_base0 + e * 16, s + e * 16);
        }
    }
    wait_async0();
    __syncthreads();

    v8f acc0 = {}, acc1 = {};

    for (int c = 0; c < 5; ++c) {
        // prefetch next chunk into the other LDS buffer (overlaps WMMAs below)
        if (c < 4) {
            const char* s = (const char*)(BpBlk + (size_t)(c + 1) * CHUNK * 512);
            unsigned int d = ((c + 1) & 1) ? shb_base1 : shb_base0;
            #pragma unroll
            for (int q = 0; q < 4; ++q) {
                int e = q * 128 + tid;
                async_b128(d + e * 16, s + e * 16);
            }
        }

        const unsigned short* bsh = &shB[c & 1][lane * 16];

        if (c < 4) {
            // state part: pre-packed A fragments, contiguous 32 B/lane loads
            #pragma unroll
            for (int u = 0; u < CHUNK; ++u) {
                v16bf a = *reinterpret_cast<const v16bf*>(Ar); Ar += 512;
                v16bf b = *reinterpret_cast<const v16bf*>(bsh + u * 512);
                if (u & 1)
                    acc1 = __builtin_amdgcn_wmma_f32_16x16x32_bf16(
                        false, a, false, b, (short)0, acc1, false, false);
                else
                    acc0 = __builtin_amdgcn_wmma_f32_16x16x32_bf16(
                        false, a, false, b, (short)0, acc0, false, false);
            }
        } else {
            // input part: build A fragment from X on the fly.
            // 16-bit A layout: lane (row m) holds runs k = 8*half + [0..7]
            // and k = 16 + 8*half + [0..7] per 32-K tile.
            #pragma unroll
            for (int u = 0; u < CHUNK; ++u) {
                const float* p = px + u * 32;
                float4 f0 = *reinterpret_cast<const float4*>(p);
                float4 f1 = *reinterpret_cast<const float4*>(p + 4);
                float4 g0 = *reinterpret_cast<const float4*>(p + 16);
                float4 g1 = *reinterpret_cast<const float4*>(p + 20);
                union { unsigned short s[16]; v16bf v; } ax;
                ax.s[0]  = f32_to_bf16_rne(f0.x); ax.s[1]  = f32_to_bf16_rne(f0.y);
                ax.s[2]  = f32_to_bf16_rne(f0.z); ax.s[3]  = f32_to_bf16_rne(f0.w);
                ax.s[4]  = f32_to_bf16_rne(f1.x); ax.s[5]  = f32_to_bf16_rne(f1.y);
                ax.s[6]  = f32_to_bf16_rne(f1.z); ax.s[7]  = f32_to_bf16_rne(f1.w);
                ax.s[8]  = f32_to_bf16_rne(g0.x); ax.s[9]  = f32_to_bf16_rne(g0.y);
                ax.s[10] = f32_to_bf16_rne(g0.z); ax.s[11] = f32_to_bf16_rne(g0.w);
                ax.s[12] = f32_to_bf16_rne(g1.x); ax.s[13] = f32_to_bf16_rne(g1.y);
                ax.s[14] = f32_to_bf16_rne(g1.z); ax.s[15] = f32_to_bf16_rne(g1.w);
                v16bf b = *reinterpret_cast<const v16bf*>(bsh + u * 512);
                if (u & 1)
                    acc1 = __builtin_amdgcn_wmma_f32_16x16x32_bf16(
                        false, ax.v, false, b, (short)0, acc1, false, false);
                else
                    acc0 = __builtin_amdgcn_wmma_f32_16x16x32_bf16(
                        false, ax.v, false, b, (short)0, acc0, false, false);
            }
        }

        // own async copies landed + all waves done reading buffer c&1
        wait_async0();
        __syncthreads();
    }

    // ---- epilogue: leaky update, write fp32 outputs + repack bf16 state ----
    // C layout: VGPR r holds (M = r + 8*half, N = nn).
    #pragma unroll
    for (int r = 0; r < 8; ++r) {
        int m = tm * 16 + r + 8 * half;
        int o = tn * 16 + nn;
        float pre  = acc0[r] + acc1[r];
        float sold = Sf32[m * OUT_DIM + o];
        float snew = (1.0f - ALPHA) * sold + ALPHA * tanhf(pre);
        Sf32[m * OUT_DIM + o] = snew;
        out[((size_t)m * (T_DIM + 1) + (size_t)(t + 1)) * OUT_DIM + o] = snew;
        // scatter into A-fragment order for next step: element (row m, k = o)
        int kk = o & 31, tkp = o >> 5;
        int hh = (kk >> 3) & 1;                 // K half-split: {8-15,24-31} -> 1
        int k2 = kk - 8 * hh;                   // in {0..7} U {16..23}
        int vg = (k2 < 8) ? (k2 >> 1) : (4 + ((k2 - 16) >> 1));
        int jj = vg * 2 + (kk & 1);
        int lp = (m & 15) + 16 * hh;
        SbfWr[(((size_t)(m >> 4) * NKT_S + tkp) * 32 + lp) * 16 + jj] =
            f32_to_bf16_rne(snew);
    }
}

extern "C" void kernel_launch(void* const* d_in, const int* in_sizes, int n_in,
                              void* d_out, int out_size, void* d_ws, size_t ws_size,
                              hipStream_t stream) {
    (void)in_sizes; (void)n_in; (void)out_size; (void)ws_size;
    const float* X     = (const float*)d_in[0];   // [B, T, IN]
    const float* w_in  = (const float*)d_in[1];   // [OUT, IN]
    const float* w_res = (const float*)d_in[2];   // [OUT, OUT]
    float* out = (float*)d_out;                   // [B, T+1, OUT]

    char* ws = (char*)d_ws;
    unsigned short* Bp   = (unsigned short*)(ws + BP_OFF);
    unsigned short* Sbf0 = (unsigned short*)(ws + SBF0_OFF);
    unsigned short* Sbf1 = (unsigned short*)(ws + SBF1_OFF);
    float*          Sf32 = (float*)(ws + SF32_OFF);
    unsigned short* sbuf[2] = { Sbf0, Sbf1 };

    esn_init<<<1024, 256, 0, stream>>>(Sf32, Sbf0, out);
    esn_pack_b<<<dim3(80, 64), 256, 0, stream>>>(w_res, w_in, Bp);

    for (int t = 0; t < T_DIM; ++t) {
        esn_step<<<256, 128, 0, stream>>>(X, Bp, sbuf[t & 1], sbuf[(t + 1) & 1],
                                          Sf32, out, t);
    }
}
